// STPGNN_51780125720692
// MI455X (gfx1250) — matrix operand order repaired and checked by reading, not compiled, not run on hardware
//
#include <hip/hip_runtime.h>

// ---------------------------------------------------------------------------
// STPGNN forward on MI455X (gfx1250, wave32).
// Heavy GEMMs (graph diffusion, adjacency construction, end MLP) run on
// v_wmma_f32_16x16x32_bf16 (bf16 in, f32 accumulate). Glue ops are f32 VALU.
// GEMM inner loop is branch-free (templated subtile count) so accumulators
// stay pinned in the WMMA register ranges (no v_mov shuffling).
// ---------------------------------------------------------------------------

typedef __bf16 v16bf __attribute__((ext_vector_type(16)));
typedef float  v8f   __attribute__((ext_vector_type(8)));

namespace {
constexpr int  kB   = 64;
constexpr int  kN   = 716;
constexpr int  kNP  = 736;   // K pad for 716 (23*32)
constexpr int  kNR  = 720;   // row pad for 716 (45*16)
constexpr int  kMG  = 384;   // max gcn   M = 32*Tout
constexpr int  kMX  = 416;   // max pgcn  M = 32*Tin
constexpr long kAdjS = (long)kNR * kNP;
constexpr long kGS   = (long)kMG * kNP;
constexpr long kXS   = (long)kMX * kNP;
constexpr int  kSK  = 416;   // total skip channels
constexpr int  kEC  = 512;
constexpr long kRows = (long)kB * kN;  // 45824, multiple of 128
}

__device__ __forceinline__ __bf16 f2bf(float f) {
  unsigned u = __float_as_uint(f);
  unsigned r = (u + 0x7FFFu + ((u >> 16) & 1u)) >> 16;
  unsigned short us = (unsigned short)r;
  return __builtin_bit_cast(__bf16, us);
}
__device__ __forceinline__ float bf2f(__bf16 h) {
  unsigned short us = __builtin_bit_cast(unsigned short, h);
  return __uint_as_float(((unsigned)us) << 16);
}

// ---------------------------------------------------------------------------
// bf16 WMMA GEMM:  C[R x S] = A[R x K] * B[K x S], B given as BT[S x K]
// (K-contiguous).  A rows padded/zeroed to R_alloc (mult of 16), BT rows to
// S_alloc (mult of 16), K multiple of 32.  f32 output.
// Block = 128 threads = 4 waves.  Each wave computes NRT (1..2) 16-row tiles
// x NJ (1..4) 16-col tiles; NRT/NJ are compile-time so the K-loop is
// straight-line: 2*NRT b128 A loads, 2*NJ b128 B loads, NRT*NJ WMMAs.
// ---------------------------------------------------------------------------
template <int NRT, int NJ>
__device__ __forceinline__ void gemm_tile(
    const __bf16* __restrict__ A, int lda,
    const __bf16* __restrict__ BT, int ldb,
    float* __restrict__ C, int ldc,
    int r0, int c0, int K, int R_store, int S_store, int ln, int hi)
{
  v8f acc[NRT][NJ];
#pragma unroll
  for (int ri = 0; ri < NRT; ++ri)
#pragma unroll
    for (int j = 0; j < NJ; ++j)
#pragma unroll
      for (int e = 0; e < 8; ++e) acc[ri][j][e] = 0.f;

  const __bf16* arow[NRT];
#pragma unroll
  for (int ri = 0; ri < NRT; ++ri)
    arow[ri] = A + (long)(r0 + 64 * ri + ln) * lda;

  for (int k0 = 0; k0 < K; k0 += 32) {
    // A fragment (16x32, ISA layout): elems 0..7 -> K = k0+8*hi+e,
    //                                 elems 8..15 -> K = k0+16+8*hi+(e-8)
    union { uint4 u[2]; v16bf v; } af[NRT];
#pragma unroll
    for (int ri = 0; ri < NRT; ++ri) {
      af[ri].u[0] = *reinterpret_cast<const uint4*>(arow[ri] + k0 + 8 * hi);
      af[ri].u[1] = *reinterpret_cast<const uint4*>(arow[ri] + k0 + 16 + 8 * hi);
    }
#pragma unroll
    for (int j = 0; j < NJ; ++j) {
      // B fragment (32x16): lane -> N, elems 0..7 -> K = k0+8*hi+e, elems 8..15 -> K = k0+16+8*hi+(e-8)
      const __bf16* brow = BT + (long)(c0 + 16 * j + ln) * ldb + k0 + 8 * hi;
      union { uint4 u[2]; v16bf v; } bfr;
      bfr.u[0] = *reinterpret_cast<const uint4*>(brow);
      bfr.u[1] = *reinterpret_cast<const uint4*>(brow + 16);
#pragma unroll
      for (int ri = 0; ri < NRT; ++ri)
        acc[ri][j] = __builtin_amdgcn_wmma_f32_16x16x32_bf16(
            false, af[ri].v, false, bfr.v, (short)0, acc[ri][j], false, false);
    }
  }

#pragma unroll
  for (int ri = 0; ri < NRT; ++ri)
#pragma unroll
    for (int j = 0; j < NJ; ++j) {
      int col = c0 + 16 * j + ln;
      if (col < S_store) {
#pragma unroll
        for (int e = 0; e < 8; ++e) {
          int row = r0 + 64 * ri + e + 8 * hi;  // D layout: VGPR e, M = e+8*hi
          if (row < R_store) C[(long)row * ldc + col] = acc[ri][j][e];
        }
      }
    }
}

__global__ __launch_bounds__(128) void gemm_bf16_kernel(
    const __bf16* __restrict__ A, long strideA, int lda,
    const __bf16* __restrict__ BT, long strideBT, int ldb,
    float* __restrict__ C, long strideC, int ldc,
    int R_store, int R_alloc, int S_store, int S_alloc, int K)
{
  int b = blockIdx.z;
  A  += (long)b * strideA;
  BT += (long)b * strideBT;
  C  += (long)b * strideC;

  int wv   = threadIdx.x >> 5;
  int lane = threadIdx.x & 31;
  int hi   = lane >> 4;
  int ln   = lane & 15;
  int r0   = blockIdx.y * 128 + wv * 16;
  int c0   = blockIdx.x * 64;
  if (r0 >= R_alloc) return;                 // wave-uniform
  int nj = (S_alloc - c0) >> 4;
  if (nj > 4) nj = 4;
  bool two = (r0 + 64) < R_alloc;            // wave-uniform

  if (two) {
    switch (nj) {
      case 1:  gemm_tile<2, 1>(A, lda, BT, ldb, C, ldc, r0, c0, K, R_store, S_store, ln, hi); break;
      case 2:  gemm_tile<2, 2>(A, lda, BT, ldb, C, ldc, r0, c0, K, R_store, S_store, ln, hi); break;
      case 3:  gemm_tile<2, 3>(A, lda, BT, ldb, C, ldc, r0, c0, K, R_store, S_store, ln, hi); break;
      default: gemm_tile<2, 4>(A, lda, BT, ldb, C, ldc, r0, c0, K, R_store, S_store, ln, hi); break;
    }
  } else {
    switch (nj) {
      case 1:  gemm_tile<1, 1>(A, lda, BT, ldb, C, ldc, r0, c0, K, R_store, S_store, ln, hi); break;
      case 2:  gemm_tile<1, 2>(A, lda, BT, ldb, C, ldc, r0, c0, K, R_store, S_store, ln, hi); break;
      case 3:  gemm_tile<1, 3>(A, lda, BT, ldb, C, ldc, r0, c0, K, R_store, S_store, ln, hi); break;
      default: gemm_tile<1, 4>(A, lda, BT, ldb, C, ldc, r0, c0, K, R_store, S_store, ln, hi); break;
    }
  }
}

// ---------------------------------------------------------------------------
// prep: inputs (pad t=0), x0 = start conv, xa = starta conv
// ---------------------------------------------------------------------------
__global__ __launch_bounds__(256) void prep_kernel(
    const float* __restrict__ hist, const float* __restrict__ sW,
    const float* __restrict__ sb, const float* __restrict__ saW,
    const float* __restrict__ sab, float* __restrict__ x0,
    float* __restrict__ xa)
{
  long total = (long)kB * kN * 13;
  for (long idx = blockIdx.x * 256L + threadIdx.x; idx < total;
       idx += (long)gridDim.x * 256) {
    int t = (int)(idx % 13);
    long r = idx / 13;
    int n = (int)(r % kN);
    int b = (int)(r / kN);
    float inp = (t == 0) ? 0.f : hist[(((long)b * 12 + (t - 1)) * kN + n) * 2];
    xa[idx] = saW[0] * inp + sab[0];
    for (int c = 0; c < 32; ++c)
      x0[(((long)b * 32 + c) * kN + n) * 13 + t] = sW[c] * inp + sb[c];
  }
}

// tmp1[b,j,k] = sum_i p1[ind[b],i] * pk[i,j,k]
__global__ __launch_bounds__(256) void tmp1_kernel(
    const float* __restrict__ hist, const float* __restrict__ p1,
    const float* __restrict__ pk, float* __restrict__ tmp1)
{
  long total = (long)kB * 40 * 40;
  for (long idx = blockIdx.x * 256L + threadIdx.x; idx < total;
       idx += (long)gridDim.x * 256) {
    int k = (int)(idx % 40);
    int j = (int)((idx / 40) % 40);
    int b = (int)(idx / 1600);
    float h = hist[(((long)b * 12 + 11) * kN + 0) * 2 + 1];
    int ind = ((int)(h * 288.f)) % 288;
    if (ind < 0) ind += 288;
    float s = 0.f;
    for (int i = 0; i < 40; ++i) s += p1[ind * 40 + i] * pk[(i * 40 + j) * 40 + k];
    tmp1[idx] = s;
  }
}

// tmp2 -> bf16 A operand [B][720][64] (k pad 40->64, rows 716->720 zero)
__global__ __launch_bounds__(256) void tmp2_pack_kernel(
    const float* __restrict__ p2, const float* __restrict__ tmp1,
    __bf16* __restrict__ A2)
{
  long total = (long)kB * kNR * 64;
  for (long idx = blockIdx.x * 256L + threadIdx.x; idx < total;
       idx += (long)gridDim.x * 256) {
    int kk = (int)(idx % 64);
    int n = (int)((idx / 64) % kNR);
    int b = (int)(idx / (64L * kNR));
    float v = 0.f;
    if (n < kN && kk < 40) {
      for (int j = 0; j < 40; ++j) v += p2[n * 40 + j] * tmp1[((long)b * 40 + j) * 40 + kk];
    }
    A2[idx] = f2bf(v);
  }
}

// generic weight -> bf16 with zero padding (row-major contiguous src)
__global__ __launch_bounds__(256) void pack_w_bf16_kernel(
    __bf16* __restrict__ dst, const float* __restrict__ src,
    int ro, int co, int ri, int ci)
{
  long total = (long)ro * co;
  for (long i = blockIdx.x * 256L + threadIdx.x; i < total;
       i += (long)gridDim.x * 256) {
    int c = (int)(i % co), r = (int)(i / co);
    float v = (r < ri && c < ci) ? src[(long)r * ci + c] : 0.f;
    dst[i] = f2bf(v);
  }
}

// per-row relu->softmax over 716 cols, write bf16 row of 736 (pad 0);
// rows >= 716 fully zeroed. Optional node mask (mask[row]*mask[col]).
__global__ __launch_bounds__(256) void relu_softmax_bf16_kernel(
    const float* __restrict__ src, long srcStride, int ldSrc,
    __bf16* __restrict__ dst, long dstStride,
    const unsigned char* __restrict__ mask)
{
  int b = blockIdx.y;
  int row = blockIdx.x;
  const float* s = src + (long)b * srcStride + (long)row * ldSrc;
  __bf16* d = dst + (long)b * dstStride + (long)row * kNP;
  int tid = threadIdx.x;
  __shared__ float red[256];
  if (row >= kN) {
    for (int c = tid; c < kNP; c += 256) d[c] = f2bf(0.f);
    return;
  }
  float rowm = (mask == nullptr || mask[row]) ? 1.f : 0.f;
  float mx = 0.f;
  for (int c = tid; c < kN; c += 256) {
    float v = s[c];
    if (mask) v *= rowm * (mask[c] ? 1.f : 0.f);
    mx = fmaxf(mx, fmaxf(v, 0.f));
  }
  red[tid] = mx; __syncthreads();
  for (int st = 128; st > 0; st >>= 1) { if (tid < st) red[tid] = fmaxf(red[tid], red[tid + st]); __syncthreads(); }
  mx = red[0]; __syncthreads();
  float sum = 0.f;
  for (int c = tid; c < kN; c += 256) {
    float v = s[c];
    if (mask) v *= rowm * (mask[c] ? 1.f : 0.f);
    sum += __expf(fmaxf(v, 0.f) - mx);
  }
  red[tid] = sum; __syncthreads();
  for (int st = 128; st > 0; st >>= 1) { if (tid < st) red[tid] += red[tid + st]; __syncthreads(); }
  float inv = 1.f / red[0];
  for (int c = tid; c < kNP; c += 256) {
    float o = 0.f;
    if (c < kN) {
      float v = s[c];
      if (mask) v *= rowm * (mask[c] ? 1.f : 0.f);
      o = __expf(fmaxf(v, 0.f) - mx) * inv;
    }
    d[c] = f2bf(o);
  }
}

// xs[n,t] = sum_b xa[b,n,t]
__global__ void xs_kernel(const float* __restrict__ xa, float* __restrict__ xs)
{
  int idx = blockIdx.x * blockDim.x + threadIdx.x;
  if (idx >= kN * 13) return;
  float s = 0.f;
  for (int b = 0; b < kB; ++b) s += xa[(long)b * kN * 13 + idx];
  xs[idx] = s;
}

// adjp[i,j] = (sum_t xs[i,t]*xs[j,t+1]) / y[j] * pivw[i,j]
__global__ __launch_bounds__(256) void adjp_kernel(
    const float* __restrict__ xs, const float* __restrict__ pivw,
    float* __restrict__ adjp)
{
  long total = (long)kN * kN;
  for (long idx = blockIdx.x * 256L + threadIdx.x; idx < total;
       idx += (long)gridDim.x * 256) {
    int j = (int)(idx % kN), i = (int)(idx / kN);
    float y = 0.f;
    for (int t = 0; t < 13; ++t) y += xs[j * 13 + t];
    float s = 0.f;
    for (int t = 0; t < 12; ++t) s += xs[i * 13 + t] * xs[j * 13 + t + 1];
    adjp[idx] = s / y * pivw[idx];
  }
}

__global__ void score_kernel(const float* __restrict__ adjp, float* __restrict__ score)
{
  int k = blockIdx.x * blockDim.x + threadIdx.x;
  if (k >= kN) return;
  float s = 0.f;
  for (int i = 0; i < kN; ++i) s += adjp[(long)i * kN + k] + adjp[(long)k * kN + i];
  score[k] = s;
}

// top-k (k=100) mask via rank counting (ties broken by index, as lax.top_k)
__global__ __launch_bounds__(256) void topk_mask_kernel(
    const float* __restrict__ score, unsigned char* __restrict__ mask)
{
  __shared__ float s[kN];
  for (int i = threadIdx.x; i < kN; i += blockDim.x) s[i] = score[i];
  __syncthreads();
  for (int i = threadIdx.x; i < kN; i += blockDim.x) {
    float si = s[i];
    int cnt = 0;
    for (int j = 0; j < kN; ++j) {
      float sj = s[j];
      cnt += (sj > si) || (sj == si && j < i);
    }
    mask[i] = (cnt < 100) ? 1 : 0;
  }
}

// gated dilated conv: fg = tanh(convF)*sigmoid(convG), kernel 1x2, dilation d
__global__ __launch_bounds__(256) void gated_conv_kernel(
    const float* __restrict__ x, float* __restrict__ fg,
    const float* __restrict__ Wf, const float* __restrict__ bfp,
    const float* __restrict__ Wg, const float* __restrict__ bgp,
    int Tin, int d)
{
  __shared__ float sWf[2048], sWg[2048], sbf[32], sbg[32];
  for (int i = threadIdx.x; i < 2048; i += 256) { sWf[i] = Wf[i]; sWg[i] = Wg[i]; }
  if (threadIdx.x < 32) { sbf[threadIdx.x] = bfp[threadIdx.x]; sbg[threadIdx.x] = bgp[threadIdx.x]; }
  __syncthreads();
  int Tout = Tin - d;
  long total = (long)kB * 32 * kN * Tout;
  for (long idx = blockIdx.x * 256L + threadIdx.x; idx < total;
       idx += (long)gridDim.x * 256) {
    int t = (int)(idx % Tout);
    long r = idx / Tout;
    int n = (int)(r % kN); r /= kN;
    int o = (int)(r % 32);
    int b = (int)(r / 32);
    float af = sbf[o], ag = sbg[o];
    const float* xb = x + ((long)b * 32 * kN + n) * Tin + t;
    for (int i2 = 0; i2 < 32; ++i2) {
      float x0 = xb[(long)i2 * kN * Tin];
      float x1 = xb[(long)i2 * kN * Tin + d];
      af += sWf[(o * 32 + i2) * 2] * x0 + sWf[(o * 32 + i2) * 2 + 1] * x1;
      ag += sWg[(o * 32 + i2) * 2] * x0 + sWg[(o * 32 + i2) * 2 + 1] * x1;
    }
    fg[idx] = tanhf(af) * (1.f / (1.f + __expf(-ag)));
  }
}

// [B,32,716,T] f32  ->  bf16 BT operand [M=32T rows][736] (v pad 0)
__global__ __launch_bounds__(256) void pack_xt_kernel(
    __bf16* __restrict__ dst, long dstStride, const float* __restrict__ src, int T)
{
  int b = blockIdx.z, m = blockIdx.y;
  int c = m / T, t = m % T;
  const float* s = src + ((long)b * 32 + c) * kN * (long)T + t;
  __bf16* d = dst + (long)b * dstStride + (long)m * kNP;
  for (int v = threadIdx.x; v < kNP; v += 256)
    d[v] = f2bf(v < kN ? s[(long)v * T] : 0.f);
}

// C f32 [716 x ldSrc] -> bf16 CT [M rows][736]  (CT[m][v] = C[v][m])
__global__ __launch_bounds__(256) void transpose_pack_kernel(
    __bf16* __restrict__ dst, long dstStride,
    const float* __restrict__ src, long srcStride, int ldSrc)
{
  int b = blockIdx.z, m = blockIdx.y;
  const float* s = src + (long)b * srcStride + m;
  __bf16* d = dst + (long)b * dstStride + (long)m * kNP;
  for (int v = threadIdx.x; v < kNP; v += 256)
    d[v] = f2bf(v < kN ? s[(long)v * ldSrc] : 0.f);
}

// fused: gconv(1x1 over [x,x1,x2]) for gcn & pgcn, alpha mix, residual add,
// write pre-BN x_new, and skip 1x1 conv into skip_t[b][n][416]
__global__ __launch_bounds__(128) void combine_kernel(
    const __bf16* __restrict__ FGT, const __bf16* __restrict__ C1T,
    const __bf16* __restrict__ C2T, const __bf16* __restrict__ XT,
    const __bf16* __restrict__ P1T, const __bf16* __restrict__ P2T,
    const float* __restrict__ xres, float* __restrict__ xnew,
    float* __restrict__ skip_t,
    const float* __restrict__ gW, const float* __restrict__ gB,
    const float* __restrict__ pW, const float* __restrict__ pB,
    const float* __restrict__ sW, const float* __restrict__ sB,
    const float* __restrict__ alpha, int Tin, int d, int skipOff)
{
  int Tout = Tin - d;
  __shared__ float sWg[3072], sWp[3072], sWs[256], sbG[32], sbP[32], sbS[8];
  for (int i = threadIdx.x; i < 3072; i += 128) { sWg[i] = gW[i]; sWp[i] = pW[i]; }
  for (int i = threadIdx.x; i < 256; i += 128) sWs[i] = sW[i];
  if (threadIdx.x < 32) { sbG[threadIdx.x] = gB[threadIdx.x]; sbP[threadIdx.x] = pB[threadIdx.x]; }
  if (threadIdx.x < 8) sbS[threadIdx.x] = sB[threadIdx.x];
  __syncthreads();
  float a_s = 1.f / (1.f + __expf(-alpha[0]));
  long total = (long)kB * kN * Tout;
  for (long idx = blockIdx.x * 128L + threadIdx.x; idx < total;
       idx += (long)gridDim.x * 128) {
    int t = (int)(idx % Tout);
    long r = idx / Tout;
    int v = (int)(r % kN);
    int b = (int)(r / kN);
    const __bf16* g0 = FGT + (long)b * kGS;
    const __bf16* g1 = C1T + (long)b * kGS;
    const __bf16* g2 = C2T + (long)b * kGS;
    const __bf16* a0 = XT + (long)b * kXS;
    const __bf16* a1 = P1T + (long)b * kXS;
    const __bf16* a2 = P2T + (long)b * kXS;
    float accg[32], acca[32];
    for (int o = 0; o < 32; ++o) { accg[o] = sbG[o]; acca[o] = sbP[o]; }
    for (int c = 0; c < 32; ++c) {
      long mg = (long)(c * Tout + t) * kNP + v;
      long mx = (long)(c * Tin + t) * kNP + v;
      float gx = bf2f(g0[mg]), g1v = bf2f(g1[mg]), g2v = bf2f(g2[mg]);
      float ax = bf2f(a0[mx]), a1v = bf2f(a1[mx]), a2v = bf2f(a2[mx]);
      for (int o = 0; o < 32; ++o) {
        accg[o] += sWg[o * 96 + c] * gx + sWg[o * 96 + 32 + c] * g1v + sWg[o * 96 + 64 + c] * g2v;
        acca[o] += sWp[o * 96 + c] * ax + sWp[o * 96 + 32 + c] * a1v + sWp[o * 96 + 64 + c] * a2v;
      }
    }
    float ss[8];
    for (int sc = 0; sc < 8; ++sc) ss[sc] = sbS[sc];
    for (int o = 0; o < 32; ++o) {
      float res = xres[(((long)b * 32 + o) * kN + v) * Tin + t + d];
      float xv = a_s * acca[o] + (1.f - a_s) * accg[o] + res;
      xnew[(((long)b * 32 + o) * kN + v) * Tout + t] = xv;
      for (int sc = 0; sc < 8; ++sc) ss[sc] += sWs[sc * 32 + o] * xv;
    }
    for (int sc = 0; sc < 8; ++sc)
      skip_t[((long)b * kN + v) * kSK + skipOff + sc * Tout + t] = ss[sc];
  }
}

// BatchNorm: block-per-channel reduction (mean,var), then elementwise apply
__global__ __launch_bounds__(256) void bn_reduce_kernel(
    const float* __restrict__ x, float* __restrict__ stats, int Tout)
{
  int c = blockIdx.x;
  long per = (long)kN * Tout;
  long n = (long)kB * per;
  float s = 0.f, s2 = 0.f;
  for (long i = threadIdx.x; i < n; i += 256) {
    long b = i / per, r = i % per;
    float v = x[(b * 32 + c) * per + r];
    s += v; s2 += v * v;
  }
  __shared__ float r1[256], r2[256];
  r1[threadIdx.x] = s; r2[threadIdx.x] = s2; __syncthreads();
  for (int st = 128; st > 0; st >>= 1) {
    if (threadIdx.x < st) { r1[threadIdx.x] += r1[threadIdx.x + st]; r2[threadIdx.x] += r2[threadIdx.x + st]; }
    __syncthreads();
  }
  if (threadIdx.x == 0) {
    float mean = r1[0] / (float)n;
    stats[c] = mean;
    stats[32 + c] = r2[0] / (float)n - mean * mean;
  }
}

__global__ __launch_bounds__(256) void bn_apply_kernel(
    const float* __restrict__ xin, float* __restrict__ xout,
    const float* __restrict__ stats, const float* __restrict__ gamma,
    const float* __restrict__ beta, int Tout)
{
  long per = (long)kN * Tout;
  long total = (long)kB * 32 * per;
  for (long i = blockIdx.x * 256L + threadIdx.x; i < total;
       i += (long)gridDim.x * 256) {
    int c = (int)((i / per) % 32);
    float mean = stats[c], var = stats[32 + c];
    xout[i] = (xin[i] - mean) * rsqrtf(var + 1e-5f) * gamma[c] + beta[c];
  }
}

// f32 -> bf16 with optional per-column bias and relu (contiguous rows x cols)
__global__ __launch_bounds__(256) void bias_relu_bf16_kernel(
    __bf16* __restrict__ dst, const float* __restrict__ src,
    const float* __restrict__ bias, long rows, int cols)
{
  long total = rows * (long)cols;
  for (long i = blockIdx.x * 256L + threadIdx.x; i < total;
       i += (long)gridDim.x * 256) {
    int c = (int)(i % cols);
    float v = src[i] + (bias ? bias[c] : 0.f);
    dst[i] = f2bf(fmaxf(v, 0.f));
  }
}

// scatter final GEMM output [(b,n) x 16] -> d_out [B,12,716], add end2 bias
__global__ __launch_bounds__(256) void final_out_kernel(
    float* __restrict__ out, const float* __restrict__ o2,
    const float* __restrict__ b2)
{
  long total = (long)kB * kN * 12;
  for (long i = blockIdx.x * 256L + threadIdx.x; i < total;
       i += (long)gridDim.x * 256) {
    int o = (int)(i % 12);
    long r = i / 12;
    int n = (int)(r % kN);
    int b = (int)(r / kN);
    out[((long)b * 12 + o) * kN + n] = o2[((long)b * kN + n) * 16 + o] + b2[o];
  }
}

// ---------------------------------------------------------------------------
extern "C" void kernel_launch(void* const* d_in, const int* in_sizes, int n_in,
                              void* d_out, int out_size, void* d_ws, size_t ws_size,
                              hipStream_t stream) {
  const float* hist  = (const float*)d_in[0];
  const float* alpha = (const float*)d_in[1];
  const float* p1    = (const float*)d_in[2];
  const float* p2    = (const float*)d_in[3];
  const float* p3    = (const float*)d_in[4];
  const float* pk    = (const float*)d_in[5];
  const float* pivw  = (const float*)d_in[6];
  const float* stW   = (const float*)d_in[7];
  const float* stB   = (const float*)d_in[8];
  const float* saW   = (const float*)d_in[9];
  const float* saB   = (const float*)d_in[10];
  const float* fW    = (const float*)d_in[11];
  const float* fB    = (const float*)d_in[12];
  const float* gW    = (const float*)d_in[13];
  const float* gB    = (const float*)d_in[14];
  const float* skW   = (const float*)d_in[15];
  const float* skB   = (const float*)d_in[16];
  const float* gcW   = (const float*)d_in[17];
  const float* gcB   = (const float*)d_in[18];
  const float* pgW   = (const float*)d_in[19];
  const float* pgB   = (const float*)d_in[20];
  const float* bnG   = (const float*)d_in[21];
  const float* bnBt  = (const float*)d_in[22];
  const float* e1W   = (const float*)d_in[23];
  const float* e1B   = (const float*)d_in[24];
  const float* e2W   = (const float*)d_in[25];
  const float* e2B   = (const float*)d_in[26];
  float* outp = (float*)d_out;
  (void)in_sizes; (void)n_in; (void)out_size; (void)ws_size;

  char* ws = (char*)d_ws;
  size_t off = 0;
  auto alloc = [&](size_t bytes) -> char* {
    char* p = ws + off;
    off += (bytes + 255) & ~(size_t)255;
    return p;
  };

  float*  xA     = (float*)alloc((size_t)kB * 32 * kN * 13 * 4);
  float*  xB     = (float*)alloc((size_t)kB * 32 * kN * 13 * 4);
  float*  xtmp   = (float*)alloc((size_t)kB * 32 * kN * 12 * 4);
  float*  fgbuf  = (float*)alloc((size_t)kB * 32 * kN * 12 * 4);
  float*  xabuf  = (float*)alloc((size_t)kB * kN * 13 * 4);
  __bf16* adjBF  = (__bf16*)alloc((size_t)kB * kAdjS * 2);
  __bf16* adjPBF = (__bf16*)alloc((size_t)kAdjS * 2);
  float*  Craw   = (float*)alloc((size_t)kB * kAdjS * 4);   // also end-GEMM out
  __bf16* FGT    = (__bf16*)alloc((size_t)kB * kGS * 2);
  __bf16* C1T    = (__bf16*)alloc((size_t)kB * kGS * 2);
  __bf16* C2T    = (__bf16*)alloc((size_t)kB * kGS * 2);
  __bf16* XT     = (__bf16*)alloc((size_t)kB * kXS * 2);
  __bf16* P1T    = (__bf16*)alloc((size_t)kB * kXS * 2);
  __bf16* P2T    = (__bf16*)alloc((size_t)kB * kXS * 2);
  float*  skipT  = (float*)alloc((size_t)kRows * kSK * 4);
  __bf16* bfbuf  = (__bf16*)alloc((size_t)kRows * kEC * 2); // skipBF then O1BF
  float*  tmp1   = (float*)alloc((size_t)kB * 1600 * 4);
  __bf16* A2     = (__bf16*)alloc((size_t)kB * kNR * 64 * 2);
  __bf16* p3BF   = (__bf16*)alloc((size_t)768 * 64 * 2);
  __bf16* e1BF   = (__bf16*)alloc((size_t)kEC * kSK * 2);
  __bf16* e2BF   = (__bf16*)alloc((size_t)16 * kEC * 2);
  float*  adjp   = (float*)alloc((size_t)kN * kN * 4);
  float*  xsb    = (float*)alloc((size_t)kN * 13 * 4);
  float*  scoreb = (float*)alloc((size_t)kN * 4);
  unsigned char* maskb = (unsigned char*)alloc(kN);
  float*  bnst   = (float*)alloc(64 * 4);

  auto nb = [](long total, int bs) { return (unsigned)((total + bs - 1) / bs); };

  auto gemm = [&](const __bf16* A, long sA, int lda, const __bf16* BT, long sBT,
                  int ldb, float* C, long sC, int ldc, int Rst, int Ral,
                  int Sst, int Sal, int K, int batch) {
    dim3 g((Sal + 63) / 64, (Ral + 127) / 128, batch);
    gemm_bf16_kernel<<<g, 128, 0, stream>>>(A, sA, lda, BT, sBT, ldb, C, sC,
                                            ldc, Rst, Ral, Sst, Sal, K);
  };

  // ---- prep ----
  prep_kernel<<<nb((long)kB * kN * 13, 256), 256, 0, stream>>>(
      hist, stW, stB, saW, saB, xA, xabuf);

  // ---- dynamic adjacency ----
  tmp1_kernel<<<nb((long)kB * 1600, 256), 256, 0, stream>>>(hist, p1, pk, tmp1);
  tmp2_pack_kernel<<<nb((long)kB * kNR * 64, 256), 256, 0, stream>>>(p2, tmp1, A2);
  pack_w_bf16_kernel<<<nb(768L * 64, 256), 256, 0, stream>>>(p3BF, p3, 768, 64, kN, 40);
  gemm(A2, (long)kNR * 64, 64, p3BF, 0, 64, Craw, kAdjS, kNP,
       kN, kNR, kN, 768, 64, kB);
  relu_softmax_bf16_kernel<<<dim3(kNR, kB), 256, 0, stream>>>(
      Craw, kAdjS, kNP, adjBF, kAdjS, nullptr);

  // ---- pivotal adjacency ----
  xs_kernel<<<nb(kN * 13L, 256), 256, 0, stream>>>(xabuf, xsb);
  adjp_kernel<<<nb((long)kN * kN, 256), 256, 0, stream>>>(xsb, pivw, adjp);
  score_kernel<<<nb(kN, 256), 256, 0, stream>>>(adjp, scoreb);
  topk_mask_kernel<<<1, 256, 0, stream>>>(scoreb, maskb);
  relu_softmax_bf16_kernel<<<dim3(kNR, 1), 256, 0, stream>>>(
      adjp, 0, kN, adjPBF, 0, maskb);

  // ---- layers ----
  static const int TSEQ[9] = {13, 12, 10, 9, 7, 6, 4, 3, 1};
  static const int DIL[8]  = {1, 2, 1, 2, 1, 2, 1, 2};
  static const int SOFF[8] = {320, 240, 168, 112, 64, 32, 8, 0};

  float* xcur = xA;
  float* xnext = xB;
  for (int l = 0; l < 8; ++l) {
    int Tin = TSEQ[l], d = DIL[l], Tout = Tin - d;
    int Mg = 32 * Tout, Mx = 32 * Tin;

    gated_conv_kernel<<<nb((long)kB * 32 * kN * Tout, 256), 256, 0, stream>>>(
        xcur, fgbuf, fW + l * 2048, fB + l * 32, gW + l * 2048, gB + l * 32, Tin, d);

    pack_xt_kernel<<<dim3(1, Mg, kB), 256, 0, stream>>>(FGT, kGS, fgbuf, Tout);
    pack_xt_kernel<<<dim3(1, Mx, kB), 256, 0, stream>>>(XT, kXS, xcur, Tin);

    // gcn diffusion: x1 = A*x, x2 = A*x1
    gemm(adjBF, kAdjS, kNP, FGT, kGS, kNP, Craw, kAdjS, kMX,
         kN, kNR, Mg, Mg, kNP, kB);
    transpose_pack_kernel<<<dim3(1, Mg, kB), 256, 0, stream>>>(C1T, kGS, Craw, kAdjS, kMX);
    gemm(adjBF, kAdjS, kNP, C1T, kGS, kNP, Craw, kAdjS, kMX,
         kN, kNR, Mg, Mg, kNP, kB);
    transpose_pack_kernel<<<dim3(1, Mg, kB), 256, 0, stream>>>(C2T, kGS, Craw, kAdjS, kMX);

    // pgcn diffusion on residual with shared pivotal adjacency
    gemm(adjPBF, 0, kNP, XT, kXS, kNP, Craw, kAdjS, kMX,
         kN, kNR, Mx, Mx, kNP, kB);
    transpose_pack_kernel<<<dim3(1, Mx, kB), 256, 0, stream>>>(P1T, kXS, Craw, kAdjS, kMX);
    gemm(adjPBF, 0, kNP, P1T, kXS, kNP, Craw, kAdjS, kMX,
         kN, kNR, Mx, Mx, kNP, kB);
    transpose_pack_kernel<<<dim3(1, Mx, kB), 256, 0, stream>>>(P2T, kXS, Craw, kAdjS, kMX);

    combine_kernel<<<nb((long)kB * kN * Tout, 128), 128, 0, stream>>>(
        FGT, C1T, C2T, XT, P1T, P2T, xcur, xtmp, skipT,
        gcW + l * 3072, gcB + l * 32, pgW + l * 3072, pgB + l * 32,
        skW + l * 256, skB + l * 8, alpha, Tin, d, SOFF[l]);

    bn_reduce_kernel<<<32, 256, 0, stream>>>(xtmp, bnst, Tout);
    bn_apply_kernel<<<nb((long)kB * 32 * kN * Tout, 256), 256, 0, stream>>>(
        xtmp, xnext, bnst, bnG + l * 32, bnBt + l * 32, Tout);

    float* t = xcur; xcur = xnext; xnext = t;
  }

  // ---- end MLP:  relu(skip) -> end1 -> relu -> end2 ----
  bias_relu_bf16_kernel<<<nb(kRows * (long)kSK, 256), 256, 0, stream>>>(
      bfbuf, skipT, nullptr, kRows, kSK);                       // skipBF
  pack_w_bf16_kernel<<<nb((long)kEC * kSK, 256), 256, 0, stream>>>(
      e1BF, e1W, kEC, kSK, kEC, kSK);
  pack_w_bf16_kernel<<<nb(16L * kEC, 256), 256, 0, stream>>>(
      e2BF, e2W, 16, kEC, 12, kEC);

  gemm(bfbuf, 0, kSK, e1BF, 0, kSK, Craw, 0, kEC,
       (int)kRows, (int)kRows, kEC, kEC, kSK, 1);
  bias_relu_bf16_kernel<<<nb(kRows * (long)kEC, 256), 256, 0, stream>>>(
      bfbuf, Craw, e1B, kRows, kEC);                            // O1BF
  gemm(bfbuf, 0, kEC, e2BF, 0, kEC, Craw, 0, 16,
       (int)kRows, (int)kRows, 12, 16, kEC, 1);

  final_out_kernel<<<nb((long)kB * kN * 12, 256), 256, 0, stream>>>(outp, Craw, e2B);
}
